// DogDetector_4827543241460
// MI455X (gfx1250) — matrix operand order repaired and checked by reading, not compiled, not run on hardware
//
#include <hip/hip_runtime.h>
#include <math.h>

// ---------------------------------------------------------------------------
// DogDetector (ResNet18 backbone + heads + NMS) for MI455X / gfx1250.
// - All conv compute on v_wmma_f32_16x16x32_f16 (f16 in, f32 acc).
// - Weights converted fp32->fp16 once per launch; ALL activations stored f16
//   (halves HBM traffic; model weights are L2-resident on the 192MB L2).
// - Weight tiles staged into LDS with GLOBAL_LOAD_ASYNC_TO_LDS_B64 (ASYNCcnt).
// - BN folded into per-channel scale/bias; bias/residual/ReLU fused in the
//   conv epilogue.  Workspace requirement: ~120 MB.
// ---------------------------------------------------------------------------

typedef __attribute__((ext_vector_type(16))) _Float16 v16h;
typedef __attribute__((ext_vector_type(8)))  float    v8f;
typedef __attribute__((__vector_size__(8)))  int      a2i;   // async b64 payload

#define TILE_M 64
#define TILE_N 64
#define TILE_K 32
#define KPAD   36   // LDS row stride (f16 elems): 72B rows dodge bank conflicts

#if defined(__gfx1250__) && __has_builtin(__builtin_amdgcn_global_load_async_to_lds_b64) && __has_builtin(__builtin_amdgcn_s_wait_asynccnt)
#define HAVE_ASYNC_LDS 1
#else
#define HAVE_ASYNC_LDS 0
#endif

// ---------------------------------------------------------------------------
// Implicit-GEMM conv:  Y[n,co,oy,ox] = sum_{ci,ky,kx} W[co,ci,ky,kx]*X[n,ci,iy,ix]
// GEMM view: M=Cout, N=B*OH*OW, K=Cin*KH*KW.
// 256 threads = 8 wave32 waves arranged 2(M)x4(N); each wave owns a 32x16
// output sub-tile = two 16x16 f32 accumulators.
// ---------------------------------------------------------------------------
__global__ __launch_bounds__(256) void conv_igemm_wmma(
    const _Float16* __restrict__ X, const _Float16* __restrict__ Wt,
    _Float16* __restrict__ Y,
    const float* __restrict__ scale, const float* __restrict__ bias,
    const _Float16* __restrict__ resid, int relu,
    int B, int Cin, int H, int Win, int Cout, int OH, int OW,
    int KH, int KW, int stride, int pad)
{
  __shared__ _Float16 As[TILE_M][KPAD];   // [m][k]
  __shared__ _Float16 Bs[TILE_N][KPAD];   // [j][k]  (im2col^T: k contiguous)
  __shared__ int jn[TILE_N], jy[TILE_N], jx[TILE_N];    // column -> (n,oy,ox)
  __shared__ int kci[TILE_K], kky[TILE_K], kkx[TILE_K]; // k -> (ci,ky,kx)

  const int K     = Cin * KH * KW;
  const int Ngemm = B * OH * OW;
  const int tid   = threadIdx.x;
  const int lane  = tid & 31;
  const int wave  = tid >> 5;
  const int wm    = wave >> 2;   // 0..1 : which 32-row M chunk
  const int wn    = wave & 3;    // 0..3 : which 16-col N chunk
  const int l16   = lane & 15;
  const int halfL = lane >> 4;

  const int mBlk = blockIdx.y * TILE_M;
  const int nBlk = blockIdx.x * TILE_N;

  if (tid < TILE_N) {
    int gj = nBlk + tid;
    if (gj < Ngemm) {
      int n = gj / (OH * OW);
      int r = gj - n * (OH * OW);
      int oy = r / OW;
      jn[tid] = n; jy[tid] = oy; jx[tid] = r - oy * OW;
    } else {
      jn[tid] = -1; jy[tid] = 0; jx[tid] = 0;
    }
  }

  const bool aAligned = ((K & 3) == 0) && (mBlk + TILE_M <= Cout);

  v8f acc0 = {};
  v8f acc1 = {};

  for (int k0 = 0; k0 < K; k0 += TILE_K) {
    if (tid < TILE_K) {
      int gk = k0 + tid;
      if (gk < K) {
        int ci = gk / (KH * KW);
        int r  = gk - ci * (KH * KW);
        int ky = r / KW;
        kci[tid] = ci; kky[tid] = ky; kkx[tid] = r - ky * KW;
      } else {
        kci[tid] = -1; kky[tid] = 0; kkx[tid] = 0;
      }
    }
    __syncthreads();   // decode tables ready; previous fragment reads done

    // ---- stage A: f16 weight tile ----
    const bool fullA = aAligned && (k0 + TILE_K <= K);
#if HAVE_ASYNC_LDS
    if (fullA) {
      // 512 b64 chunks (4 halfs each); rows are 8B-aligned (K%4==0, KPAD*2=72B)
#pragma unroll
      for (int it = 0; it < 2; ++it) {
        int t  = tid + it * 256;
        int m  = t >> 3;
        int kk = (t & 7) << 2;
        __builtin_amdgcn_global_load_async_to_lds_b64(
            (__attribute__((address_space(1))) a2i*)(uintptr_t)(
                Wt + (size_t)(mBlk + m) * K + k0 + kk),
            (__attribute__((address_space(3))) a2i*)(uintptr_t)(&As[m][kk]),
            0, 0);
      }
    } else
#endif
    {
      for (int t = tid; t < TILE_M * TILE_K; t += 256) {
        int m  = t >> 5;
        int kk = t & 31;
        int gm = mBlk + m;
        int gk = k0 + kk;
        _Float16 v = (_Float16)0.0f;
        if (gm < Cout && gk < K) v = Wt[(size_t)gm * K + gk];
        As[m][kk] = v;
      }
      // prefetch next weight K-chunk (global_prefetch_b8)
      if (k0 + TILE_K < K && tid < TILE_M) {
        int gm = mBlk + tid;
        if (gm < Cout) __builtin_prefetch(&Wt[(size_t)gm * K + k0 + TILE_K], 0, 1);
      }
    }
    // ---- stage B: im2col gather (f16 activations) ----
    for (int t = tid; t < TILE_N * TILE_K; t += 256) {
      int j  = t >> 5;
      int kk = t & 31;
      _Float16 v = (_Float16)0.0f;
      int n  = jn[j];
      int ci = kci[kk];
      if (n >= 0 && ci >= 0) {
        int iy = jy[j] * stride - pad + kky[kk];
        int ix = jx[j] * stride - pad + kkx[kk];
        if (iy >= 0 && iy < H && ix >= 0 && ix < Win)
          v = X[((size_t)(n * Cin + ci) * H + iy) * Win + ix];
      }
      Bs[j][kk] = v;
    }
#if HAVE_ASYNC_LDS
    __builtin_amdgcn_s_wait_asynccnt(0);
#endif
    __syncthreads();

    // ---- assemble fragments per ISA 7.12.2 layouts and issue WMMA ----
    v16h a0, a1, bfr;
#pragma unroll
    for (int v = 0; v < 8; ++v) {
      // A 16x32 (16-bit): v<4 -> K = 2v + 8*half ; v>=4 -> K = 16 + 2(v-4) + 8*half
      int kb = (v < 4) ? (2 * v + 8 * halfL) : (16 + 2 * (v - 4) + 8 * halfL);
      int mA = wm * 32 + l16;
      a0[2 * v]     = As[mA][kb];
      a0[2 * v + 1] = As[mA][kb + 1];
      a1[2 * v]     = As[mA + 16][kb];
      a1[2 * v + 1] = As[mA + 16][kb + 1];
      // B 32x16 (16-bit): same K order as A, N = lane%16
      int kbB = kb;
      int jB  = wn * 16 + l16;
      bfr[2 * v]     = Bs[jB][kbB];
      bfr[2 * v + 1] = Bs[jB][kbB + 1];
    }
    acc0 = __builtin_amdgcn_wmma_f32_16x16x32_f16(false, a0, false, bfr,
                                                  (short)0, acc0, false, false);
    acc1 = __builtin_amdgcn_wmma_f32_16x16x32_f16(false, a1, false, bfr,
                                                  (short)0, acc1, false, false);
  }

  // ---- epilogue: reuse LDS column tables; fused scale/bias/residual/relu ----
  {
    int jB = wn * 16 + l16;
    int n  = jn[jB];
    if (n >= 0) {
      int oy = jy[jB];
      int ox = jx[jB];
#pragma unroll
      for (int t = 0; t < 2; ++t) {
#pragma unroll
        for (int v = 0; v < 8; ++v) {
          int co = mBlk + wm * 32 + t * 16 + v + 8 * halfL;
          if (co < Cout) {
            float y = t ? acc1[v] : acc0[v];
            if (scale)      y = y * scale[co] + bias[co];
            else if (bias)  y += bias[co];
            size_t off = ((size_t)(n * Cout + co) * OH + oy) * OW + ox;
            if (resid) y += (float)resid[off];
            if (relu)  y = fmaxf(y, 0.0f);
            Y[off] = (_Float16)y;
          }
        }
      }
    }
  }
}

// ---------------------------------------------------------------------------
// fp32 -> fp16 conversion (weights + network input, once per launch).
// ---------------------------------------------------------------------------
__global__ void f32_to_f16_k(const float* __restrict__ src,
                             _Float16* __restrict__ dst, int n)
{
  int i = blockIdx.x * blockDim.x + threadIdx.x;
  if (i < n) dst[i] = (_Float16)src[i];
}

// ---------------------------------------------------------------------------
// Fold eval-mode BN (+ optional conv bias) into per-channel scale/bias.
// ---------------------------------------------------------------------------
__global__ void bn_fold_k(const float* __restrict__ g, const float* __restrict__ b,
                          const float* __restrict__ m, const float* __restrict__ v,
                          const float* __restrict__ convb,
                          float* __restrict__ scale, float* __restrict__ bias, int C)
{
  int i = blockIdx.x * blockDim.x + threadIdx.x;
  if (i >= C) return;
  float s  = g[i] * rsqrtf(v[i] + 1e-5f);
  float cb = convb ? convb[i] : 0.0f;
  scale[i] = s;
  bias[i]  = (cb - m[i]) * s + b[i];
}

// ---------------------------------------------------------------------------
// 3x3 stride-2 pad-1 maxpool (NCHW, f16 in/out, f32 compare).
// ---------------------------------------------------------------------------
__global__ void maxpool3x3s2(const _Float16* __restrict__ X, _Float16* __restrict__ Y,
                             int B, int C, int H, int W, int OH, int OW)
{
  int idx   = blockIdx.x * blockDim.x + threadIdx.x;
  int total = B * C * OH * OW;
  if (idx >= total) return;
  int ox = idx % OW;
  int t  = idx / OW;
  int oy = t % OH; t /= OH;
  int c  = t % C;
  int n  = t / C;
  float m = -3.4e38f;
  int y0 = oy * 2 - 1, x0 = ox * 2 - 1;
#pragma unroll
  for (int dy = 0; dy < 3; ++dy)
#pragma unroll
    for (int dx = 0; dx < 3; ++dx) {
      int iy = y0 + dy, ix = x0 + dx;
      if (iy >= 0 && iy < H && ix >= 0 && ix < W)
        m = fmaxf(m, (float)X[((size_t)(n * C + c) * H + iy) * W + ix]);
    }
  Y[idx] = (_Float16)m;
}

// ---------------------------------------------------------------------------
// Anchor generation + box decode + sigmoid.  441 anchors = 7x7 cells x 9.
// ---------------------------------------------------------------------------
__global__ void decode_k(const _Float16* __restrict__ braw,
                         const _Float16* __restrict__ craw,
                         float* __restrict__ boxes, float* __restrict__ conf, int B)
{
  int idx   = blockIdx.x * blockDim.x + threadIdx.x;
  int total = B * 441;
  if (idx >= total) return;
  int b = idx / 441;
  int a = idx % 441;
  int i = a / 63;        // 63 = 7 * 9
  int r = a % 63;
  int j = r / 9;
  int c = r % 9;
  const float S[3] = {0.3f, 0.5f, 0.7f};
  const float R[3] = {0.7f, 1.0f, 1.3f};
  float s  = S[c / 3];
  float rr = R[c % 3];
  float sq = sqrtf(rr);
  float aw = s * sq;
  float ah = s / sq;
  float acx = (j + 0.5f) / 7.0f;
  float acy = (i + 0.5f) / 7.0f;

  float t0 = (float)braw[((b * 36 + (c * 4 + 0)) * 7 + i) * 7 + j];
  float t1 = (float)braw[((b * 36 + (c * 4 + 1)) * 7 + i) * 7 + j];
  float t2 = (float)braw[((b * 36 + (c * 4 + 2)) * 7 + i) * 7 + j];
  float t3 = (float)braw[((b * 36 + (c * 4 + 3)) * 7 + i) * 7 + j];

  float pcx = t0 * aw + acx;
  float pcy = t1 * ah + acy;
  float psw = expf(fminf(t2, 4.0f)) * aw;
  float psh = expf(fminf(t3, 4.0f)) * ah;

  float* bp = boxes + (size_t)idx * 4;
  bp[0] = pcx - psw * 0.5f;
  bp[1] = pcy - psh * 0.5f;
  bp[2] = pcx + psw * 0.5f;
  bp[3] = pcy + psh * 0.5f;

  float lc = (float)craw[((b * 9 + c) * 7 + i) * 7 + j];
  conf[idx] = 1.0f / (1.0f + expf(-lc));
}

// ---------------------------------------------------------------------------
// Greedy padded NMS, one block per image, all state in LDS.
// Output layout (floats): kb[B][10][4] | ks[B][10] | kv[B][10]
// ---------------------------------------------------------------------------
#define NANCH 441
#define MAXDET 10
__global__ __launch_bounds__(256) void nms_k(const float* __restrict__ boxes,
                                             const float* __restrict__ conf,
                                             float* __restrict__ out, int B)
{
  __shared__ float sx1[NANCH], sy1[NANCH], sx2[NANCH], sy2[NANCH], sar[NANCH], sw[NANCH];
  __shared__ float rv[256];
  __shared__ int   ri[256];
  int b   = blockIdx.x;
  int tid = threadIdx.x;

  for (int a = tid; a < NANCH; a += 256) {
    const float* bp = boxes + ((size_t)b * NANCH + a) * 4;
    float x1 = fminf(fmaxf(bp[0], 0.0f), 1.0f);
    float y1 = fminf(fmaxf(bp[1], 0.0f), 1.0f);
    float x2 = fminf(fmaxf(bp[2], 0.0f), 1.0f);
    float y2 = fminf(fmaxf(bp[3], 0.0f), 1.0f);
    sx1[a] = x1; sy1[a] = y1; sx2[a] = x2; sy2[a] = y2;
    sar[a] = (x2 - x1) * (y2 - y1);
    float s = conf[b * NANCH + a];
    sw[a] = (s > 0.5f) ? s : -1.0f;
  }
  __syncthreads();

  for (int det = 0; det < MAXDET; ++det) {
    float bv = -1e30f;
    int   bi = 0;
    for (int a = tid; a < NANCH; a += 256)
      if (sw[a] > bv) { bv = sw[a]; bi = a; }
    rv[tid] = bv; ri[tid] = bi;
    __syncthreads();
    for (int off = 128; off > 0; off >>= 1) {
      if (tid < off) {
        float ov = rv[tid + off];
        int   oi = ri[tid + off];
        if (ov > rv[tid] || (ov == rv[tid] && oi < ri[tid])) { rv[tid] = ov; ri[tid] = oi; }
      }
      __syncthreads();
    }
    int   j = ri[0];
    float s = rv[0];
    int valid = (s > 0.0f) ? 1 : 0;
    float jx1 = sx1[j], jy1 = sy1[j], jx2 = sx2[j], jy2 = sy2[j], jar = sar[j];
    if (valid) {
      for (int a = tid; a < NANCH; a += 256) {
        float ix1 = fmaxf(jx1, sx1[a]);
        float iy1 = fmaxf(jy1, sy1[a]);
        float ix2 = fminf(jx2, sx2[a]);
        float iy2 = fminf(jy2, sy2[a]);
        float inter = fmaxf(ix2 - ix1, 0.0f) * fmaxf(iy2 - iy1, 0.0f);
        float iou = inter / (sar[a] + jar - inter + 1e-9f);
        if (iou > 0.5f || a == j) sw[a] = -1.0f;
      }
    }
    if (tid == 0) {
      float* kb = out + ((size_t)b * MAXDET + det) * 4;
      kb[0] = valid ? jx1 : 0.0f;
      kb[1] = valid ? jy1 : 0.0f;
      kb[2] = valid ? jx2 : 0.0f;
      kb[3] = valid ? jy2 : 0.0f;
      out[(size_t)B * MAXDET * 4 + b * MAXDET + det] = valid ? s : 0.0f;
      out[(size_t)B * MAXDET * 5 + b * MAXDET + det] = valid ? 1.0f : 0.0f;
    }
    __syncthreads();
  }
}

// ---------------------------------------------------------------------------
// Host orchestration.
// Input leaves follow JAX tree_flatten order (dict keys sorted at each level):
// 0:bbox_b 1:bbox_w 2..96:blocks 97..100:bn1(b,g,m,v) 101:ch1_b 102..105:ch1_bn
// 106:ch1_w 107:ch2_b 108..111:ch2_bn 112:ch2_w 113:cls1_b 114..117:cls1_bn
// 118:cls1_w 119:cls2_b 120:cls2_w 121:conv1 122:lat_b 123:lat_w 124:sm_b
// 125:sm_w 126:x.  Per block: b1(b,g,m,v) b2(b,g,m,v) c1 c2 [dsb(b,g,m,v) dsc].
// ---------------------------------------------------------------------------
extern "C" void kernel_launch(void* const* d_in, const int* in_sizes, int n_in,
                              void* d_out, int out_size, void* d_ws, size_t ws_size,
                              hipStream_t stream)
{
  (void)in_sizes; (void)n_in; (void)out_size; (void)ws_size;
  auto F = [&](int i) -> const float* { return (const float*)d_in[i]; };

  float* ws = (float*)d_ws;
  size_t o = 0;   // in float (4-byte) slots
  auto allocH = [&](size_t halves) -> _Float16* {
    _Float16* p = (_Float16*)(ws + o);
    o += (halves + 1) / 2;
    return p;
  };
  auto allocF = [&](size_t floats) -> float* {
    float* p = ws + o;
    o += floats;
    return p;
  };

  const int B = 32;

  _Float16* xh     = allocH(4816896);    // 32*3*224*224
  _Float16* buf[3];
  buf[0] = allocH(25690112);             // 32*64*112*112 (conv1 out)
  buf[1] = allocH(6422528);              // 32*64*56*56
  buf[2] = allocH(6422528);
  _Float16* hA      = allocH(401408);    // 32*256*7*7
  _Float16* hB      = allocH(401408);
  _Float16* bboxraw = allocH(56448);     // 32*36*7*7
  _Float16* clsraw  = allocH(14112);     // 32*9*7*7
  float*    boxes   = allocF(56448);     // 32*441*4
  float*    conf    = allocF(14112);     // 32*441
  float*    fold    = allocF(16384);     // folded BN scale/bias tables
  size_t fo = 0;
  // f16 weight arena (~13.8M halves)
  _Float16* wArena  = allocH(14000000);
  size_t wo = 0;

  auto cvt = [&](const float* src, _Float16* dst, int n) {
    f32_to_f16_k<<<dim3((n + 255) / 256), dim3(256), 0, stream>>>(src, dst, n);
  };
  auto cvtW = [&](const float* src, int n) -> const _Float16* {
    _Float16* dst = wArena + wo;
    wo += (size_t)n;
    cvt(src, dst, n);
    return dst;
  };
  auto foldBN = [&](const float* g, const float* bb, const float* m, const float* v,
                    const float* convb, int C, float** sc, float** bi) {
    *sc = fold + fo;
    *bi = fold + fo + C;
    fo += 2 * (size_t)C;
    bn_fold_k<<<dim3((C + 63) / 64), dim3(64), 0, stream>>>(g, bb, m, v, convb, *sc, *bi, C);
  };
  auto conv = [&](const _Float16* X, const _Float16* Wt, _Float16* Y,
                  const float* sc, const float* bi, const _Float16* resid, int relu,
                  int Cin, int H, int Wi, int Cout, int KH, int KW, int st, int pad) {
    int OH = (H + 2 * pad - KH) / st + 1;
    int OW = (Wi + 2 * pad - KW) / st + 1;
    long Ng = (long)B * OH * OW;
    dim3 g((unsigned)((Ng + TILE_N - 1) / TILE_N), (unsigned)((Cout + TILE_M - 1) / TILE_M));
    conv_igemm_wmma<<<g, dim3(256), 0, stream>>>(X, Wt, Y, sc, bi, resid, relu,
                                                 B, Cin, H, Wi, Cout, OH, OW, KH, KW, st, pad);
  };

  // network input -> f16
  cvt(F(126), xh, 4816896);

  // conv1 (7x7 s2 p3) + bn1 + relu -> buf0 [32,64,112,112]
  {
    float *sc, *bi;
    foldBN(F(98), F(97), F(99), F(100), nullptr, 64, &sc, &bi);
    conv(xh, cvtW(F(121), 64 * 3 * 49), buf[0], sc, bi, nullptr, 1,
         3, 224, 224, 64, 7, 7, 2, 3);
  }
  // maxpool 3x3 s2 p1 -> buf1 [32,64,56,56]
  {
    int total = B * 64 * 56 * 56;
    maxpool3x3s2<<<dim3((total + 255) / 256), dim3(256), 0, stream>>>(
        buf[0], buf[1], B, 64, 112, 112, 56, 56);
  }

  // residual blocks
  static const int blk_start[8] = {2, 12, 22, 37, 47, 62, 72, 87};
  static const int spec_cin[8]  = {64, 64, 64, 128, 128, 256, 256, 512};
  static const int spec_cout[8] = {64, 64, 128, 128, 256, 256, 512, 512};
  static const int spec_s[8]    = {1, 1, 2, 1, 2, 1, 2, 1};
  int cur = 1, hin = 56;
  for (int k = 0; k < 8; ++k) {
    int bs = blk_start[k];
    int cin = spec_cin[k], cout = spec_cout[k], st = spec_s[k];
    bool ds = (st != 1) || (cin != cout);
    int hout = hin / st;
    const _Float16* c1 = cvtW(F(bs + 8), cout * cin * 9);
    const _Float16* c2 = cvtW(F(bs + 9), cout * cout * 9);
    float *s1, *b1, *s2, *b2;
    foldBN(F(bs + 1), F(bs + 0), F(bs + 2), F(bs + 3), nullptr, cout, &s1, &b1); // b1
    foldBN(F(bs + 5), F(bs + 4), F(bs + 6), F(bs + 7), nullptr, cout, &s2, &b2); // b2
    if (!ds) {
      int f1 = (cur + 1) % 3;
      conv(buf[cur], c1, buf[f1], s1, b1, nullptr, 1, cin, hin, hin, cout, 3, 3, st, 1);
      // conv2 + bn2 + residual(in-place) + relu
      conv(buf[f1], c2, buf[cur], s2, b2, buf[cur], 1, cout, hout, hout, cout, 3, 3, 1, 1);
    } else {
      int f1 = (cur + 1) % 3, f2 = (cur + 2) % 3;
      float *sd, *bd;
      foldBN(F(bs + 11), F(bs + 10), F(bs + 12), F(bs + 13), nullptr, cout, &sd, &bd); // dsb
      const _Float16* dscw = cvtW(F(bs + 14), cout * cin);
      conv(buf[cur], dscw, buf[f2], sd, bd, nullptr, 0, cin, hin, hin, cout, 1, 1, st, 0);
      conv(buf[cur], c1, buf[f1], s1, b1, nullptr, 1, cin, hin, hin, cout, 3, 3, st, 1);
      conv(buf[f1], c2, buf[f2], s2, b2, buf[f2], 1, cout, hout, hout, cout, 3, 3, 1, 1);
      cur = f2;
    }
    hin = hout;
  }

  // heads (all 7x7 spatial)
  conv(buf[cur], cvtW(F(123), 256 * 512), hA, nullptr, F(122), nullptr, 0,
       512, 7, 7, 256, 1, 1, 1, 0);                                                     // lat
  conv(hA, cvtW(F(125), 256 * 256 * 9), hB, nullptr, F(124), nullptr, 0,
       256, 7, 7, 256, 3, 3, 1, 1);                                                     // sm
  {
    float *sc, *bi;
    foldBN(F(103), F(102), F(104), F(105), F(101), 256, &sc, &bi);
    conv(hB, cvtW(F(106), 256 * 256 * 9), hA, sc, bi, nullptr, 1, 256, 7, 7, 256, 3, 3, 1, 1); // ch1
  }
  {
    float *sc, *bi;
    foldBN(F(109), F(108), F(110), F(111), F(107), 256, &sc, &bi);
    conv(hA, cvtW(F(112), 256 * 256 * 9), hB, sc, bi, nullptr, 1, 256, 7, 7, 256, 3, 3, 1, 1); // ch2
  }
  conv(hB, cvtW(F(1), 36 * 256 * 9), bboxraw, nullptr, F(0), nullptr, 0,
       256, 7, 7, 36, 3, 3, 1, 1);                                                      // bbox
  {
    float *sc, *bi;
    foldBN(F(115), F(114), F(116), F(117), F(113), 256, &sc, &bi);
    conv(hB, cvtW(F(118), 256 * 256 * 9), hA, sc, bi, nullptr, 1, 256, 7, 7, 256, 3, 3, 1, 1); // cls1
  }
  conv(hA, cvtW(F(120), 9 * 256 * 9), clsraw, nullptr, F(119), nullptr, 0,
       256, 7, 7, 9, 3, 3, 1, 1);                                                       // cls2

  // decode + NMS
  {
    int total = B * 441;
    decode_k<<<dim3((total + 255) / 256), dim3(256), 0, stream>>>(bboxraw, clsraw, boxes, conf, B);
  }
  nms_k<<<dim3(B), dim3(256), 0, stream>>>(boxes, conf, (float*)d_out, B);
}
